// GraphAttentionNetworkMH_27986006901172
// MI455X (gfx1250) — matrix operand order repaired and checked by reading, not compiled, not run on hardware
//
#include <hip/hip_runtime.h>

#define Bdim 4
#define Kdim 256
#define Ddim 128
#define Hdim 4

typedef __attribute__((ext_vector_type(16))) __bf16 v16bf;
typedef __attribute__((ext_vector_type(8)))  __bf16 v8bf;
typedef __attribute__((ext_vector_type(8)))  float  v8f;
typedef __attribute__((ext_vector_type(4)))  unsigned int u32x4;
typedef __attribute__((ext_vector_type(8)))  int i32x8;
typedef __attribute__((ext_vector_type(4)))  int i32x4;
typedef unsigned short ushort_t;

#if defined(__HIP_DEVICE_COMPILE__) && __has_builtin(__builtin_amdgcn_tensor_load_to_lds)
#define HAVE_TDM 1
#else
#define HAVE_TDM 0
#endif

__device__ __forceinline__ unsigned short f2bf(float f) {
  union { float f; unsigned u; } v; v.f = f;
  unsigned u = v.u;
  return (unsigned short)((u + 0x7FFFu + ((u >> 16) & 1u)) >> 16);
}

__device__ __forceinline__ v16bf join8(v8bf lo, v8bf hi) {
  v16bf r;
#pragma unroll
  for (int i = 0; i < 8; ++i) { r[i] = lo[i]; r[8 + i] = hi[i]; }
  return r;
}

// A-fragment (16x32 bf16) loader from row-major bf16 memory.
// Lane layout: lane L holds row L&15; K = {hi*16..hi*16+15} (contiguous, matches B loads).
__device__ __forceinline__ v16bf load_a_frag(const ushort_t* rowbase, int kk, int laneHi) {
  v8bf lo = *(const v8bf*)(rowbase + kk * 32 + laneHi * 16);
  v8bf hi = *(const v8bf*)(rowbase + kk * 32 + laneHi * 16 + 8);
  return join8(lo, hi);
}

#if HAVE_TDM
// Issue one TDM load of a contiguous 2D block (rows x rowDwords dwords, packed)
// into LDS at byte offset lds_off. D# layout per CDNA5 ISA.
__device__ __forceinline__ void tdm_load_2d(unsigned lds_off, const void* gptr,
                                            unsigned rowDwords, unsigned rows) {
  unsigned long long ga = (unsigned long long)gptr;
  u32x4 g0;
  g0[0] = 1u;                                              // count=1, no gather
  g0[1] = lds_off;                                         // lds_addr
  g0[2] = (unsigned)(ga & 0xFFFFFFFFu);                    // global_addr[31:0]
  g0[3] = (unsigned)((ga >> 32) & 0x01FFFFFFu) | (2u << 30); // addr[56:32], type=2
  i32x8 g1;
  g1[0] = (int)(2u << 16);                                 // wg_mask=0, data_size=2 (4B)
  g1[1] = (int)((rowDwords & 0xFFFFu) << 16);              // tensor_dim0[15:0]
  g1[2] = (int)(((rowDwords >> 16) & 0xFFFFu) |            // tensor_dim0[31:16]
                ((rows & 0xFFFFu) << 16));                 // tensor_dim1[15:0]
  g1[3] = (int)(((rows >> 16) & 0xFFFFu) |                 // tensor_dim1[31:16]
                ((rowDwords & 0xFFFFu) << 16));            // tile_dim0
  g1[4] = (int)(rows & 0xFFFFu);                           // tile_dim1 (tile_dim2=0)
  g1[5] = (int)rowDwords;                                  // tensor_dim0_stride[31:0]
  g1[6] = 0;                                               // stride0 hi / stride1 lo
  g1[7] = 0;
  i32x4 z4 = {0, 0, 0, 0};
#if defined(__clang_major__) && __clang_major__ >= 23
  i32x8 z8 = {0, 0, 0, 0, 0, 0, 0, 0};
  __builtin_amdgcn_tensor_load_to_lds(g0, g1, z4, z4, z8, 0);
#else
  __builtin_amdgcn_tensor_load_to_lds(g0, g1, z4, z4, 0);
#endif
}
#endif

// ---------------------------------------------------------------------------
// Kernel A: f32 -> bf16 conversions.
//   xbf  [b][k][d]   row-major          (A-operand for GEMM 1)
//   xbfT [b][d][k]   transposed         (B-operand for attn @ x)
//   wt   [h][c][d]   wt[h][c][d] = W[h][(c>=128?128:0)+d][c%128]  (B-operand GEMM 1)
// ---------------------------------------------------------------------------
__global__ __launch_bounds__(256)
void k_convert(const float* __restrict__ x, const float* __restrict__ W,
               ushort_t* __restrict__ xbf, ushort_t* __restrict__ xbfT,
               ushort_t* __restrict__ wt) {
  int idx = blockIdx.x * blockDim.x + threadIdx.x;  // 0 .. 131071
  {
    int d = idx & (Ddim - 1);
    int k = (idx >> 7) & (Kdim - 1);
    int b = idx >> 15;
    unsigned short v = f2bf(x[idx]);
    xbf[idx] = v;
    xbfT[(b * Ddim + d) * Kdim + k] = v;
  }
  {
    int q = idx & (Ddim - 1);
    int p = (idx >> 7) & 255;
    int h = idx >> 15;
    int part = p >> 7;
    int dd = p & 127;
    int c = part * Ddim + q;
    wt[(h * 256 + c) * Ddim + dd] = f2bf(W[idx]);
  }
}

// ---------------------------------------------------------------------------
// Kernel B: LR GEMM via WMMA. Per (h,b): out[k,c] = sum_d x[k,d]*wt[h][c][d].
//   c <  128: Lbuf[h][b][k][c]            (left, row-major)
//   c >= 128: RcT [h][b][c-128][k]        (right + lin_b, TRANSPOSED)
// ---------------------------------------------------------------------------
__global__ __launch_bounds__(256)
void k_gemm_lr(const ushort_t* __restrict__ xbf, const ushort_t* __restrict__ wt,
               const float* __restrict__ lin_b,
               float* __restrict__ Lbuf, float* __restrict__ RcT) {
  const int mtile = blockIdx.x;               // 0..3 -> 64 rows each
  const int b = blockIdx.y, h = blockIdx.z;
  const int tid = threadIdx.x;
  const int wave = tid >> 5, lane = tid & 31;
  const int ln = lane & 15, hi = lane >> 4;

  const ushort_t* xb = xbf + b * Kdim * Ddim;
  const ushort_t* wh = wt + h * 256 * Ddim;
  float* Lhb = Lbuf + (h * Bdim + b) * Kdim * Ddim;
  float* Rhb = RcT + (h * Bdim + b) * Ddim * Kdim;

  for (int t = wave; t < 64; t += 8) {        // 4 row-tiles x 16 col-tiles
    int mt = t >> 4, nt = t & 15;
    int row0 = mtile * 64 + mt * 16;
    v8f acc = {};
    const ushort_t* arow = xb + (row0 + ln) * Ddim;
    const ushort_t* brow = wh + (nt * 16 + ln) * Ddim;
    // prefetch next tile's B operand row (emits global_prefetch_b8)
    {
      int t2 = t + 8;
      if (t2 < 64) __builtin_prefetch(wh + ((t2 & 15) * 16 + ln) * Ddim, 0, 1);
    }
#pragma unroll
    for (int kk = 0; kk < 4; ++kk) {          // Kdim inner = 128 = 4*32
      v16bf af = load_a_frag(arow, kk, hi);
      v16bf bf = *(const v16bf*)(brow + kk * 32 + hi * 16);
      acc = __builtin_amdgcn_wmma_f32_16x16x32_bf16(false, af, false, bf,
                                                    (short)0, acc, false, false);
    }
    int gc = nt * 16 + ln;                    // output column
    int rbase = row0 + hi * 8;                // output row base for this lane
    if (gc < Ddim) {
#pragma unroll
      for (int r = 0; r < 8; ++r) Lhb[(rbase + r) * Ddim + gc] = acc[r];
    } else {
      int dc = gc - Ddim;
      float lb = lin_b[h * Ddim + dc];
#pragma unroll
      for (int r = 0; r < 8; ++r) Rhb[dc * Kdim + (rbase + r)] = acc[r] + lb;
    }
  }
}

// ---------------------------------------------------------------------------
// Kernel C: fused e / softmax / attn@x per (h, b, 32-row tile).
// LDS: RcT (128x256 f32), xT (128x256 bf16), attn (32x256 bf16),
//      Lrows (32x128 f32), a (128 f32), dotRc (256), dotL (32).
// e_ij = 0.6*(dotL_i + dotRc_j) + 0.4*sum_d |L_id + Rc_jd| * a_d + bias_ij
// ---------------------------------------------------------------------------
#define SM_RCT   0
#define SM_XT    (SM_RCT + 128 * 256 * 4)      /* 131072 */
#define SM_ATTN  (SM_XT + 128 * 256 * 2)       /* 196608 */
#define SM_LROWS (SM_ATTN + 32 * 256 * 2)      /* 212992 */
#define SM_A     (SM_LROWS + 32 * 128 * 4)     /* 229376 */
#define SM_DOTRC (SM_A + 128 * 4)              /* 229888 */
#define SM_DOTL  (SM_DOTRC + 256 * 4)          /* 230912 */
#define SM_TOTAL (SM_DOTL + 32 * 4)            /* 231040 bytes < 320KB */

__global__ __launch_bounds__(256)
void k_attn(const float* __restrict__ a_vec, const float* __restrict__ bias,
            const float* __restrict__ Lbuf, const float* __restrict__ RcT,
            const ushort_t* __restrict__ xbfT, float* __restrict__ Obuf) {
  extern __shared__ char smem[];
  float*    rct_s   = (float*)(smem + SM_RCT);
  ushort_t* xt_s    = (ushort_t*)(smem + SM_XT);
  ushort_t* attn_s  = (ushort_t*)(smem + SM_ATTN);
  float*    lrows_s = (float*)(smem + SM_LROWS);
  float*    a_s     = (float*)(smem + SM_A);
  float*    dotrc_s = (float*)(smem + SM_DOTRC);
  float*    dotl_s  = (float*)(smem + SM_DOTL);

  const int tile = blockIdx.x;                // 0..7 -> rows [tile*32, +32)
  const int b = blockIdx.y, h = blockIdx.z;
  const int tid = threadIdx.x;
  const int wave = tid >> 5, lane = tid & 31;
  const int i0 = tile * 32;

  const float* Rhb = RcT + (h * Bdim + b) * Ddim * Kdim;
  const float* Lrows_g = Lbuf + ((h * Bdim + b) * Kdim + i0) * Ddim;
  const ushort_t* xtb = xbfT + b * Ddim * Kdim;

  // ---- stage LDS -------------------------------------------------------
#if HAVE_TDM
  {
    unsigned lbase = (unsigned)(size_t)(void*)smem;  // LDS byte offset (flat addr low bits)
    if (wave == 0) {
      tdm_load_2d(lbase + SM_RCT, Rhb, 256, 128);          // 128KB f32
      __builtin_amdgcn_s_wait_tensorcnt(0);
    } else if (wave == 1) {
      tdm_load_2d(lbase + SM_XT, xtb, 128, 128);           // 64KB bf16 (as dwords)
      __builtin_amdgcn_s_wait_tensorcnt(0);
    } else if (wave == 2) {
      tdm_load_2d(lbase + SM_LROWS, Lrows_g, 4096, 1);     // 16KB f32
      __builtin_amdgcn_s_wait_tensorcnt(0);
    }
    if (tid < 128) a_s[tid] = a_vec[h * Ddim + tid];
  }
#else
  {
    const uint4* s1 = (const uint4*)Rhb;     uint4* d1 = (uint4*)rct_s;
    for (int i = tid; i < 8192; i += 256) d1[i] = s1[i];
    const uint4* s2 = (const uint4*)xtb;     uint4* d2 = (uint4*)xt_s;
    for (int i = tid; i < 4096; i += 256) d2[i] = s2[i];
    const uint4* s3 = (const uint4*)Lrows_g; uint4* d3 = (uint4*)lrows_s;
    for (int i = tid; i < 1024; i += 256) d3[i] = s3[i];
    if (tid < 128) a_s[tid] = a_vec[h * Ddim + tid];
  }
#endif
  __syncthreads();

  // ---- linear-part dot products ---------------------------------------
  {
    float s = 0.f;
#pragma unroll 4
    for (int d = 0; d < 128; ++d) s += rct_s[d * 256 + tid] * a_s[d];
    dotrc_s[tid] = s;
    if (tid < 32) {
      float t2 = 0.f;
#pragma unroll 4
      for (int d = 0; d < 128; ++d) t2 += lrows_s[tid * 128 + d] * a_s[d];
      dotl_s[tid] = t2;
    }
  }
  __syncthreads();

  // ---- e + softmax: one wave per row; lane owns j = q*32 + lane --------
  for (int il = wave; il < 32; il += 8) {
    float acc[8];
#pragma unroll
    for (int q = 0; q < 8; ++q) acc[q] = 0.f;
    const float* lr = lrows_s + il * 128;
    for (int d = 0; d < 128; ++d) {
      float Lv = lr[d];                       // LDS broadcast
      float av = a_s[d];                      // LDS broadcast
      const float* rrow = rct_s + d * 256 + lane;
#pragma unroll
      for (int q = 0; q < 8; ++q) {
        float u = Lv + rrow[q * 32];
        acc[q] = fmaf(fabsf(u), av, acc[q]);  // |u| folds into fma abs modifier
      }
    }
    float dl = dotl_s[il];
    const float* brow = bias + (h * Kdim + (i0 + il)) * Kdim;
    float e[8], m = -3.4e38f;
#pragma unroll
    for (int q = 0; q < 8; ++q) {
      int j = q * 32 + lane;
      e[q] = 0.6f * (dl + dotrc_s[j]) + 0.4f * acc[q] + brow[j];
      m = fmaxf(m, e[q]);
    }
#pragma unroll
    for (int off = 16; off >= 1; off >>= 1) m = fmaxf(m, __shfl_xor(m, off, 32));
    float s = 0.f, p[8];
#pragma unroll
    for (int q = 0; q < 8; ++q) { p[q] = __expf(e[q] - m); s += p[q]; }
#pragma unroll
    for (int off = 16; off >= 1; off >>= 1) s += __shfl_xor(s, off, 32);
    float inv = 1.0f / s;
#pragma unroll
    for (int q = 0; q < 8; ++q)
      attn_s[il * 256 + q * 32 + lane] = f2bf(p[q] * inv);
  }
  __syncthreads();

  // ---- attn(32x256) @ x(256x128) via WMMA ------------------------------
  const int ln = lane & 15, hif = lane >> 4;
  float* Ob = Obuf + ((h * Bdim + b) * Kdim + i0) * Ddim;
  for (int t = wave; t < 16; t += 8) {        // 2 M-tiles x 8 N-tiles
    int mt = t & 1, nt = t >> 1;
    v8f acc = {};
    const ushort_t* arow = attn_s + (mt * 16 + ln) * 256;
    const ushort_t* brow = xt_s + (nt * 16 + ln) * 256;  // xT[d][j], d = out col
#pragma unroll
    for (int kk = 0; kk < 8; ++kk) {          // K = 256 = 8*32
      v16bf af = load_a_frag(arow, kk, hif);
      v16bf bf = *(const v16bf*)(brow + kk * 32 + hif * 16);
      acc = __builtin_amdgcn_wmma_f32_16x16x32_bf16(false, af, false, bf,
                                                    (short)0, acc, false, false);
    }
    int dc = nt * 16 + ln;
    int rb = mt * 16 + hif * 8;
#pragma unroll
    for (int r = 0; r < 8; ++r) Ob[(rb + r) * Ddim + dc] = acc[r];
  }
}

// ---------------------------------------------------------------------------
// Kernel D: deterministic mean over heads.
// ---------------------------------------------------------------------------
__global__ __launch_bounds__(256)
void k_reduce(const float* __restrict__ Obuf, float* __restrict__ out) {
  int idx = blockIdx.x * blockDim.x + threadIdx.x;  // 0 .. 131071
  const int N = Bdim * Kdim * Ddim;
  float s = Obuf[idx] + Obuf[idx + N] + Obuf[idx + 2 * N] + Obuf[idx + 3 * N];
  out[idx] = 0.25f * s;
}

extern "C" void kernel_launch(void* const* d_in, const int* in_sizes, int n_in,
                              void* d_out, int out_size, void* d_ws, size_t ws_size,
                              hipStream_t stream) {
  (void)in_sizes; (void)n_in; (void)out_size; (void)ws_size;
  const float* x     = (const float*)d_in[0];
  const float* W     = (const float*)d_in[1];
  const float* lin_b = (const float*)d_in[2];
  const float* a     = (const float*)d_in[3];
  const float* bias  = (const float*)d_in[4];
  float* out = (float*)d_out;

  char* ws = (char*)d_ws;
  ushort_t* xbf  = (ushort_t*)(ws);                          // 256 KB
  ushort_t* xbfT = (ushort_t*)(ws + (256 << 10));            // 256 KB
  ushort_t* wt   = (ushort_t*)(ws + (512 << 10));            // 256 KB
  float* Lbuf = (float*)(ws + (768 << 10));                  // 2 MB
  float* RcT  = (float*)(ws + (768 << 10) + (2 << 20));      // 2 MB
  float* Obuf = (float*)(ws + (768 << 10) + (4 << 20));      // 2 MB

  k_convert<<<512, 256, 0, stream>>>(x, W, xbf, xbfT, wt);
  k_gemm_lr<<<dim3(4, Bdim, Hdim), 256, 0, stream>>>(xbf, wt, lin_b, Lbuf, RcT);
  k_attn<<<dim3(8, Bdim, Hdim), 256, SM_TOTAL, stream>>>(a, bias, Lbuf, RcT, xbfT, Obuf);
  k_reduce<<<512, 256, 0, stream>>>(Obuf, out);
}
